// SingleLayerRank1DCModel_37271726195000
// MI455X (gfx1250) — compile-verified, matrix operand order checked by reading, not run on hardware
//
#include <hip/hip_runtime.h>

#define T_LEN 2048
#define HID   1024
#define NH    16
#define DK    64
#define DV    128
#define VAL   2048   // NH*DV
#define QKD   1024   // NH*DK
#define CONVK 4

typedef __attribute__((ext_vector_type(16))) __bf16 v16bf;
typedef __attribute__((ext_vector_type(8)))  float  v8f;
typedef __attribute__((address_space(3))) void lds_void;

// ---------- helpers ----------
__device__ __forceinline__ unsigned short f2bf(float f) {
  unsigned u = __float_as_uint(f);
  u += 0x7FFFu + ((u >> 16) & 1u);           // round-to-nearest-even
  return (unsigned short)(u >> 16);
}

__device__ __forceinline__ float wred(float v) {
#pragma unroll
  for (int off = 16; off > 0; off >>= 1) v += __shfl_xor(v, off, 32);
  return v;
}

__device__ __forceinline__ float sigmoidf_(float x) { return 1.0f / (1.0f + __expf(-x)); }
__device__ __forceinline__ float siluf_(float x)    { return x * sigmoidf_(x); }

// ---------- CDNA5 async global -> LDS copy (16B per lane, ASYNCcnt-tracked) ----------
__device__ __forceinline__ void async_cp16(void* lds_dst, const void* gsrc) {
  asm volatile("global_load_async_to_lds_b128 %0, %1, off"
               :: "v"((lds_void*)lds_dst), "v"(gsrc)
               : "memory");
}

template <int N>
__device__ __forceinline__ void wait_async() {
#if __has_builtin(__builtin_amdgcn_s_wait_asynccnt)
  __builtin_amdgcn_s_wait_asynccnt(N);
#else
  asm volatile("s_wait_asynccnt %0" :: "i"(N) : "memory");
#endif
}

// ---------- RMSNorm: x = hs * rsqrt(mean(hs^2)+1e-6) * w ; also emit bf16 ----------
__global__ void rmsnorm_kernel(const float* __restrict__ hs, const float* __restrict__ w,
                               float* __restrict__ x, unsigned short* __restrict__ xbf) {
  const int t = blockIdx.x;
  const int tid = threadIdx.x;             // 256 threads
  const float* r = hs + (size_t)t * HID;
  __shared__ float red[8];
  __shared__ float scale;
  float s = 0.f;
  for (int i = tid; i < HID; i += 256) { float v = r[i]; s += v * v; }
  s = wred(s);
  if ((tid & 31) == 0) red[tid >> 5] = s;
  __syncthreads();
  if (tid == 0) {
    float tt = 0.f;
#pragma unroll
    for (int i = 0; i < 8; ++i) tt += red[i];
    scale = rsqrtf(tt * (1.0f / HID) + 1e-6f);
  }
  __syncthreads();
  const float sc = scale;
  for (int i = tid; i < HID; i += 256) {
    float v = r[i] * sc * w[i];
    x[(size_t)t * HID + i] = v;
    xbf[(size_t)t * HID + i] = f2bf(v);
  }
}

// ---------- f32 -> bf16 bulk convert ----------
__global__ void cvt_bf16_kernel(const float* __restrict__ in, unsigned short* __restrict__ out,
                                size_t n) {
  size_t i = (size_t)blockIdx.x * blockDim.x + threadIdx.x;
  if (i < n) out[i] = f2bf(in[i]);
}

// ---------- WMMA NT-GEMM, double-buffered async-LDS staging ----------
// C[M,N] = A[M,K] * B[N,K]^T, bf16 operands, f32 out.
// Block: 256 threads = 8 waves (4 M x 2 N), tile 64x32, K staged 64 at a time.
// Stage n+1 copies are in flight (ASYNCcnt) while stage n runs on the WMMA pipe.
__device__ __forceinline__ v16bf frag_from_lds(const unsigned short* __restrict__ base,
                                               int row0, int k0, int lane) {
  // ISA 16-bit A/B fragment layout (16x32 / 32x16), ld = 64 elements:
  //   lanes 0-15 : row = lane,    elems[0..7] = K k0..k0+7,  elems[8..15] = K k0+16..k0+23
  //   lanes 16-31: row = lane-16, elems[0..7] = K k0+8..+15, elems[8..15] = K k0+24..+31
  const int r  = row0 + (lane & 15);
  const int kb = k0 + ((lane >> 4) << 3);
  const unsigned short* p = base + r * 64 + kb;
  v16bf f;
#pragma unroll
  for (int i = 0; i < 8; ++i) {
    f[i]     = __builtin_bit_cast(__bf16, p[i]);
    f[i + 8] = __builtin_bit_cast(__bf16, p[i + 16]);
  }
  return f;
}

__global__ __launch_bounds__(256, 2)
void gemm_nt_bf16_kernel(const unsigned short* __restrict__ A,
                         const unsigned short* __restrict__ B,
                         float* __restrict__ C, int M, int N, int K) {
  __shared__ alignas(16) unsigned short sA[2][64 * 64];  // 2 x 8 KB
  __shared__ alignas(16) unsigned short sB[2][32 * 64];  // 2 x 4 KB
  const int tid  = threadIdx.x;
  const int lane = tid & 31;
  const int wid  = tid >> 5;                    // 8 waves: 4 (M) x 2 (N)
  const int m0b  = blockIdx.y * 64;
  const int n0b  = blockIdx.x * 32;
  const int wrowA = (wid >> 1) * 16;            // wave's row offset in sA
  const int wrowB = (wid & 1) * 16;             // wave's row offset in sB

  // per-thread copy coordinates: 16 bytes (8 bf16) per issue
  const int crow = tid >> 3;                    // 0..31
  const int ccol = (tid & 7) * 8;               // element offset in a 64-elem row
  const unsigned short* gA0 = A + (size_t)(m0b + crow) * K + ccol;
  const unsigned short* gA1 = A + (size_t)(m0b + crow + 32) * K + ccol;
  const unsigned short* gB  = B + (size_t)(n0b + crow) * K + ccol;
  const int ldsA = crow * 64 + ccol;
  const int ldsB = crow * 64 + ccol;

  auto issue_stage = [&](int buf, int kt) {
    async_cp16(&sA[buf][ldsA],            gA0 + kt);
    async_cp16(&sA[buf][ldsA + 32 * 64],  gA1 + kt);
    async_cp16(&sB[buf][ldsB],            gB + kt);
  };

  v8f acc = {};
  int buf = 0;
  issue_stage(0, 0);                            // prologue
  for (int kt = 0; kt < K; kt += 64) {
    const bool has_next = (kt + 64) < K;
    if (has_next) issue_stage(buf ^ 1, kt + 64);   // keep next stage in flight
    if (has_next) wait_async<3>(); else wait_async<0>();  // in-order completion
    __syncthreads();
#pragma unroll
    for (int ks = 0; ks < 64; ks += 32) {
      v16bf a = frag_from_lds(sA[buf], wrowA, ks, lane);
      v16bf b = frag_from_lds(sB[buf], wrowB, ks, lane);
      acc = __builtin_amdgcn_wmma_f32_16x16x32_bf16(
          /*neg_a=*/false, a, /*neg_b=*/false, b,
          /*c_mod=*/(short)0, acc, /*reuse_a=*/false, /*reuse_b=*/false);
    }
    __syncthreads();                            // reads of buf done before overwrite
    buf ^= 1;
  }
  // f32 C/D layout: lane n = lane&15, VGPR v -> row m = v + 8*(lane>>4)
  const int mrow = m0b + wrowA + (lane >> 4) * 8;
  const int ncol = n0b + wrowB + (lane & 15);
#pragma unroll
  for (int v = 0; v < 8; ++v) C[(size_t)(mrow + v) * N + ncol] = acc[v];
}

// ---------- a/b projections -> beta (sigmoid), g (log-decay) : one wave per (t,h) ----------
__global__ void ab_proj_kernel(const float* __restrict__ x, const float* __restrict__ a_w,
                               const float* __restrict__ b_w, const float* __restrict__ dt_bias,
                               const float* __restrict__ A_log,
                               float* __restrict__ beta, float* __restrict__ g) {
  const int gw   = (blockIdx.x * blockDim.x + threadIdx.x) >> 5;
  const int lane = threadIdx.x & 31;
  const int t = gw / NH, h = gw % NH;
  const float* xr = x + (size_t)t * HID;
  const float* ar = a_w + (size_t)h * HID;
  const float* br = b_w + (size_t)h * HID;
  float sa = 0.f, sb = 0.f;
  for (int i = lane; i < HID; i += 32) {
    float xv = xr[i];
    sa += xv * ar[i];
    sb += xv * br[i];
  }
  sa = wred(sa); sb = wred(sb);
  if (lane == 0) {
    float za = sa + dt_bias[h];
    float sp = (za > 20.f) ? za : log1pf(__expf(za));
    beta[t * NH + h] = sigmoidf_(sb);
    g[t * NH + h]    = -__expf(A_log[h]) * sp;
  }
}

// ---------- causal depthwise conv (K=4) + SiLU ----------
__global__ void conv_silu_kernel(const float* __restrict__ pre, const float* __restrict__ w,
                                 float* __restrict__ out, int C) {
  const size_t idx = (size_t)blockIdx.x * blockDim.x + threadIdx.x;
  const int t = (int)(idx / C);
  const int c = (int)(idx % C);
  float acc = 0.f;
#pragma unroll
  for (int i = 0; i < CONVK; ++i) {
    int tt = t - (CONVK - 1) + i;
    if (tt >= 0) acc += pre[(size_t)tt * C + c] * w[c * CONVK + i];
  }
  out[idx] = siluf_(acc);
}

// ---------- k DC-removal + L2 norm of q,k + q scale : one wave per (t,h) ----------
__global__ void qk_post_kernel(float* __restrict__ q, float* __restrict__ k) {
  const int gw   = (blockIdx.x * blockDim.x + threadIdx.x) >> 5;
  const int lane = threadIdx.x & 31;
  const int t = gw / NH, h = gw % NH;
  float* qr = q + ((size_t)t * NH + h) * DK;
  float* kr = k + ((size_t)t * NH + h) * DK;
  float k0 = kr[lane], k1 = kr[lane + 32];
  float q0 = qr[lane], q1 = qr[lane + 32];
  float km = wred(k0 + k1) * (1.0f / DK);
  k0 -= km; k1 -= km;
  float ks = wred(k0 * k0 + k1 * k1);
  float qs = wred(q0 * q0 + q1 * q1);
  float kn = rsqrtf(ks + 1e-6f);
  float qn = rsqrtf(qs + 1e-6f) * 0.125f;   // * DK^-0.5
  kr[lane] = k0 * kn; kr[lane + 32] = k1 * kn;
  qr[lane] = q0 * qn; qr[lane + 32] = q1 * qn;
}

// ---------- gated delta-rule recurrence: one block per head, S in registers ----------
__global__ __launch_bounds__(128, 1)
void scan_kernel(const float* __restrict__ q, const float* __restrict__ k,
                 const float* __restrict__ v, const float* __restrict__ beta,
                 const float* __restrict__ g, float* __restrict__ o) {
  const int h  = blockIdx.x;
  const int vv = threadIdx.x;            // 0..127 -> value column
  float S[DK];
#pragma unroll
  for (int i = 0; i < DK; ++i) S[i] = 0.f;
  __shared__ float sq[DK], sk[DK];
  for (int t = 0; t < T_LEN; ++t) {
    const size_t bqk = ((size_t)t * NH + h) * DK;
    if (vv < DK) { sq[vv] = q[bqk + vv]; sk[vv] = k[bqk + vv]; }
    const float vt = v[((size_t)t * NH + h) * DV + vv];
    const float bt = beta[t * NH + h];
    const float gt = g[t * NH + h];
    __syncthreads();
    const float dec = __expf(gt);
    float dot = 0.f;
#pragma unroll
    for (int i = 0; i < DK; ++i) { S[i] *= dec; dot += sk[i] * S[i]; }
    const float vn = (vt - dot) * bt;
    float out = 0.f;
#pragma unroll
    for (int i = 0; i < DK; ++i) { S[i] += sk[i] * vn; out += sq[i] * S[i]; }
    o[((size_t)t * NH + h) * DV + vv] = out;
    __syncthreads();
  }
}

// ---------- output head RMSNorm * silu(gate) -> bf16 : one wave per (t,h) ----------
__global__ void onorm_gate_kernel(const float* __restrict__ o, const float* __restrict__ gate,
                                  const float* __restrict__ onw,
                                  unsigned short* __restrict__ obf) {
  const int gw   = (blockIdx.x * blockDim.x + threadIdx.x) >> 5;
  const int lane = threadIdx.x & 31;
  const int t = gw / NH, h = gw % NH;
  const float* orr = o    + ((size_t)t * NH + h) * DV;
  const float* gr  = gate + ((size_t)t * NH + h) * DV;
  float ov[4]; float s = 0.f;
#pragma unroll
  for (int j = 0; j < 4; ++j) { ov[j] = orr[lane + 32 * j]; s += ov[j] * ov[j]; }
  s = wred(s);
  const float r = rsqrtf(s * (1.0f / DV) + 1e-5f);
#pragma unroll
  for (int j = 0; j < 4; ++j) {
    const int dv = lane + 32 * j;
    const float gv = gr[dv];
    const float val = ov[j] * r * onw[dv] * siluf_(gv);
    obf[(size_t)t * VAL + h * DV + dv] = f2bf(val);
  }
}

// ---------- launch ----------
extern "C" void kernel_launch(void* const* d_in, const int* in_sizes, int n_in,
                              void* d_out, int out_size, void* d_ws, size_t ws_size,
                              hipStream_t stream) {
  (void)in_sizes; (void)n_in; (void)out_size; (void)ws_size;
  const float* hs       = (const float*)d_in[0];
  const float* norm_w   = (const float*)d_in[1];
  const float* q_w      = (const float*)d_in[2];
  const float* k_w      = (const float*)d_in[3];
  const float* v_w      = (const float*)d_in[4];
  const float* a_w      = (const float*)d_in[5];
  const float* b_w      = (const float*)d_in[6];
  const float* g_w      = (const float*)d_in[7];
  const float* dt_bias  = (const float*)d_in[8];
  const float* A_log    = (const float*)d_in[9];
  const float* conv_q_w = (const float*)d_in[10];
  const float* conv_k_w = (const float*)d_in[11];
  const float* conv_v_w = (const float*)d_in[12];
  const float* o_norm_w = (const float*)d_in[13];
  const float* o_proj_w = (const float*)d_in[14];
  const float* out_proj_w = (const float*)d_in[15];

  char* base = (char*)d_ws;
  size_t off = 0;
  auto alloc = [&](size_t bytes) -> void* {
    void* r = base + off;
    off = (off + bytes + 255) & ~(size_t)255;
    return r;
  };

  float* x_f32 = (float*)alloc((size_t)T_LEN * HID * 4);
  unsigned short* xbf = (unsigned short*)alloc((size_t)T_LEN * HID * 2);
  unsigned short* wq  = (unsigned short*)alloc((size_t)QKD * HID * 2);
  unsigned short* wk  = (unsigned short*)alloc((size_t)QKD * HID * 2);
  unsigned short* wv  = (unsigned short*)alloc((size_t)VAL * HID * 2);
  unsigned short* wg  = (unsigned short*)alloc((size_t)VAL * HID * 2);
  unsigned short* wo  = (unsigned short*)alloc((size_t)HID * VAL * 2);
  unsigned short* wout= (unsigned short*)alloc((size_t)HID * HID * 2);
  float* q_pre  = (float*)alloc((size_t)T_LEN * QKD * 4);
  float* k_pre  = (float*)alloc((size_t)T_LEN * QKD * 4);
  float* v_pre  = (float*)alloc((size_t)T_LEN * VAL * 4);
  float* g_pre  = (float*)alloc((size_t)T_LEN * VAL * 4);
  float* qc     = (float*)alloc((size_t)T_LEN * QKD * 4);
  float* kc     = (float*)alloc((size_t)T_LEN * QKD * 4);
  float* vc     = (float*)alloc((size_t)T_LEN * VAL * 4);
  float* beta   = (float*)alloc((size_t)T_LEN * NH * 4);
  float* gdec   = (float*)alloc((size_t)T_LEN * NH * 4);
  float* o_raw  = (float*)alloc((size_t)T_LEN * VAL * 4);
  unsigned short* obf = (unsigned short*)alloc((size_t)T_LEN * VAL * 2);
  float* t1     = (float*)alloc((size_t)T_LEN * HID * 4);
  unsigned short* t1bf = (unsigned short*)alloc((size_t)T_LEN * HID * 2);

  // 1) RMSNorm (also bf16 activations)
  rmsnorm_kernel<<<T_LEN, 256, 0, stream>>>(hs, norm_w, x_f32, xbf);

  // 2) weights -> bf16
  auto cvt = [&](const float* in, unsigned short* out, size_t n) {
    cvt_bf16_kernel<<<(unsigned)((n + 255) / 256), 256, 0, stream>>>(in, out, n);
  };
  cvt(q_w,        wq,   (size_t)QKD * HID);
  cvt(k_w,        wk,   (size_t)QKD * HID);
  cvt(v_w,        wv,   (size_t)VAL * HID);
  cvt(g_w,        wg,   (size_t)VAL * HID);
  cvt(o_proj_w,   wo,   (size_t)HID * VAL);
  cvt(out_proj_w, wout, (size_t)HID * HID);

  // 3) projection GEMMs (WMMA bf16, pipelined async-LDS staging)
  dim3 blk(256, 1, 1);
  gemm_nt_bf16_kernel<<<dim3(QKD / 32, T_LEN / 64), blk, 0, stream>>>(xbf, wq, q_pre, T_LEN, QKD, HID);
  gemm_nt_bf16_kernel<<<dim3(QKD / 32, T_LEN / 64), blk, 0, stream>>>(xbf, wk, k_pre, T_LEN, QKD, HID);
  gemm_nt_bf16_kernel<<<dim3(VAL / 32, T_LEN / 64), blk, 0, stream>>>(xbf, wv, v_pre, T_LEN, VAL, HID);
  gemm_nt_bf16_kernel<<<dim3(VAL / 32, T_LEN / 64), blk, 0, stream>>>(xbf, wg, g_pre, T_LEN, VAL, HID);

  // 4) beta / g (small per-head dots)
  ab_proj_kernel<<<T_LEN * NH / 4, 128, 0, stream>>>(x_f32, a_w, b_w, dt_bias, A_log, beta, gdec);

  // 5) causal depthwise conv + SiLU
  conv_silu_kernel<<<(unsigned)((size_t)T_LEN * QKD / 256), 256, 0, stream>>>(q_pre, conv_q_w, qc, QKD);
  conv_silu_kernel<<<(unsigned)((size_t)T_LEN * QKD / 256), 256, 0, stream>>>(k_pre, conv_k_w, kc, QKD);
  conv_silu_kernel<<<(unsigned)((size_t)T_LEN * VAL / 256), 256, 0, stream>>>(v_pre, conv_v_w, vc, VAL);

  // 6) k DC-removal + L2 normalization (+ q scale)
  qk_post_kernel<<<T_LEN * NH / 4, 128, 0, stream>>>(qc, kc);

  // 7) sequential gated delta-rule scan, one block per head
  scan_kernel<<<NH, 128, 0, stream>>>(qc, kc, vc, beta, gdec, o_raw);

  // 8) gated output RMSNorm -> bf16
  onorm_gate_kernel<<<T_LEN * NH / 4, 128, 0, stream>>>(o_raw, g_pre, o_norm_w, obf);

  // 9) o_proj then out_proj (WMMA bf16)
  gemm_nt_bf16_kernel<<<dim3(HID / 32, T_LEN / 64), blk, 0, stream>>>(obf, wo, t1, T_LEN, HID, VAL);
  cvt(t1, t1bf, (size_t)T_LEN * HID);
  gemm_nt_bf16_kernel<<<dim3(HID / 32, T_LEN / 64), blk, 0, stream>>>(t1bf, wout, (float*)d_out, T_LEN, HID, HID);
}
